// SchNetEncoder_57045755625877
// MI455X (gfx1250) — matrix operand order repaired and checked by reading, not compiled, not run on hardware
//
#include <hip/hip_runtime.h>
#include <math.h>

// ---------------- problem constants (match reference) ----------------
#define NUM_NODES 50000
#define NUM_EDGES 1600000
#define NUM_GRAPHS 128
#define HID 128
#define NUM_G 50
#define CUTOFF 8.0f
#define NUM_INTER 3
#define LOG2F_ 0.6931471805599453f
#define PI_F 3.14159265358979323846f

typedef __attribute__((ext_vector_type(16))) _Float16 v16h;
typedef __attribute__((ext_vector_type(8)))  _Float16 v8h;
typedef __attribute__((ext_vector_type(8)))  float    v8f;

#define LDS_LD 136  // act tile row stride (halfs): 16x128 f16, padded -> conflict-free ds_read_b128
#define G_LD   72   // gauss tile row stride (halfs): 16x64 f16, padded, 16B-aligned rows

__device__ __forceinline__ float ssp(float x) {
  // shifted softplus: log(1+exp(x)) - log(2), stable for large x
  float sp = (x > 15.f) ? x : __logf(1.f + __expf(x));
  return sp - LOG2F_;
}

__device__ __forceinline__ v8f wmma_f16(v16h a, v16h b, v8f c) {
  return __builtin_amdgcn_wmma_f32_16x16x32_f16(false, a, false, b, (short)0, c,
                                                false, false);
}

// B fragment (32x16 f16) from row-major f32 weight W[K, HID].
// Layout (ISA 7.12.2): lane l holds column col0+(l&15); K rows (l>>4)*8 + j for j<8, +16 for j>=8.
__device__ __forceinline__ v16h load_b(const float* __restrict__ W, int Krows,
                                       int kb, int col0) {
  const int lane = threadIdx.x & 31;
  const int c = col0 + (lane & 15);
  const int kr = kb + ((lane >> 4) << 3);
  v16h b;
#pragma unroll
  for (int j = 0; j < 16; ++j) {
    const int k = kr + (j < 8 ? j : j + 8);
    b[j] = (_Float16)((k < Krows) ? W[k * HID + c] : 0.f);
  }
  return b;
}

// A fragment (16x32 f16) from global row-major f32 X[., HID], rows row0..row0+15.
// lane l = row row0+(l&15); halves 0..7 -> K = kb+(l>>4)*8+t, halves 8..15 -> +16.
__device__ __forceinline__ v16h load_a_g(const float* __restrict__ X, int row0,
                                         int kb) {
  const int lane = threadIdx.x & 31;
  const float* r = X + (size_t)(row0 + (lane & 15)) * HID;
  const int k0 = kb + ((lane >> 4) << 3);
  v16h a;
#pragma unroll
  for (int t = 0; t < 8; ++t) a[t] = (_Float16)r[k0 + t];
#pragma unroll
  for (int t = 0; t < 8; ++t) a[8 + t] = (_Float16)r[k0 + 16 + t];
  return a;
}

// A fragment from an LDS f16 tile (16 rows, row stride `ld` halfs, 16B-aligned).
__device__ __forceinline__ v16h load_a_lds(const _Float16* tile, int ld, int kb) {
  const int lane = threadIdx.x & 31;
  const _Float16* r = tile + (lane & 15) * ld;
  const int k0 = kb + ((lane >> 4) << 3);
  v8h lo = *(const v8h*)(r + k0);
  v8h hi = *(const v8h*)(r + k0 + 16);
  v16h a;
#pragma unroll
  for (int t = 0; t < 8; ++t) { a[t] = lo[t]; a[8 + t] = hi[t]; }
  return a;
}

// ---------------- kernel: edge geometry (once) ----------------
__global__ void __launch_bounds__(256)
edge_geom_kernel(const float* __restrict__ pos, const float* __restrict__ shift,
                 const int* __restrict__ src, const int* __restrict__ dst,
                 float* __restrict__ dists, float* __restrict__ cutv) {
  const int e = blockIdx.x * blockDim.x + threadIdx.x;
  if (e >= NUM_EDGES) return;
  const int s = src[e], d = dst[e];
  const float dx = pos[s * 3 + 0] - (pos[d * 3 + 0] + shift[e * 3 + 0]);
  const float dy = pos[s * 3 + 1] - (pos[d * 3 + 1] + shift[e * 3 + 1]);
  const float dz = pos[s * 3 + 2] - (pos[d * 3 + 2] + shift[e * 3 + 2]);
  const float w = sqrtf(dx * dx + dy * dy + dz * dz);
  dists[e] = w;
  cutv[e] = 0.5f * (__cosf(w * (PI_F / CUTOFF)) + 1.0f);
}

// ---------------- kernel: h = emb[z] ----------------
__global__ void __launch_bounds__(256)
init_h_kernel(const float* __restrict__ emb, const int* __restrict__ z,
              float* __restrict__ h) {
  const int i = blockIdx.x * blockDim.x + threadIdx.x;
  if (i >= NUM_NODES * HID) return;
  const int n = i >> 7, c = i & (HID - 1);
  h[i] = emb[z[n] * HID + c];
}

// ---------------- kernel: hW = h @ cf_w1 ; agg = 0 ----------------
__global__ void __launch_bounds__(256)
node_pre_kernel(const float* __restrict__ h, const float* __restrict__ cf_w1,
                float* __restrict__ hW, float* __restrict__ agg) {
  const int lane = threadIdx.x & 31;
  const int wave = threadIdx.x >> 5;
  const int col0 = wave * 16;
  const int col = col0 + (lane & 15);
  v16h b[4];
#pragma unroll
  for (int ks = 0; ks < 4; ++ks) b[ks] = load_b(cf_w1, HID, ks * 32, col0);
  const int mbase = (lane >> 4) << 3;

  for (int t = blockIdx.x; t < NUM_NODES / 16; t += gridDim.x) {
    const int row0 = t * 16;
    v8f acc = {};
#pragma unroll
    for (int ks = 0; ks < 4; ++ks)
      acc = wmma_f16(load_a_g(h, row0, ks * 32), b[ks], acc);
#pragma unroll
    for (int r = 0; r < 8; ++r) {
      const size_t idx = (size_t)(row0 + mbase + r) * HID + col;
      hW[idx] = acc[r];
      agg[idx] = 0.f;
    }
  }
}

// ---------------- kernel: per-edge filter MLP + gather-mul + scatter-add ----
// Wf = ssp(gauss(d) @ W1 + b1) @ W2 + b2, times cosine cutoff;
// agg[dst] += hW[src] * Wf.  16 edges/tile, 8 waves x 16 cols.
__global__ void __launch_bounds__(256)
edge_conv_kernel(const float* __restrict__ mlp_w1, const float* __restrict__ mlp_b1,
                 const float* __restrict__ mlp_w2, const float* __restrict__ mlp_b2,
                 const int* __restrict__ src, const int* __restrict__ dst,
                 const float* __restrict__ dists, const float* __restrict__ cutv,
                 const float* __restrict__ hW, float* __restrict__ agg) {
  __shared__ __align__(16) _Float16 sG[16 * G_LD];     // gaussian tile 16x64 (padded)
  __shared__ __align__(16) _Float16 act[16 * LDS_LD];  // activation tile 16x128
  __shared__ float sC[16];
  __shared__ int sSrc[16], sDst[16];

  const int lane = threadIdx.x & 31;
  const int wave = threadIdx.x >> 5;
  const int col0 = wave * 16;
  const int col = col0 + (lane & 15);

  // resident weight fragments for this wave's 16 output columns
  v16h b1_0 = load_b(mlp_w1, NUM_G, 0, col0);
  v16h b1_1 = load_b(mlp_w1, NUM_G, 32, col0);  // rows 50..63 zero-padded
  v16h b2[4];
#pragma unroll
  for (int ks = 0; ks < 4; ++ks) b2[ks] = load_b(mlp_w2, HID, ks * 32, col0);
  const float bias1 = mlp_b1[col];
  const float bias2 = mlp_b2[col];

  const float delta = CUTOFF / (float)(NUM_G - 1);
  const float coeff = -0.5f / (delta * delta);
  const int mbase = (lane >> 4) << 3;

  // cooperative gaussian-tile work split: each thread computes 4 K-values of 1 edge
  const int geg = threadIdx.x & 15;         // edge row 0..15
  const int gk0 = (threadIdx.x >> 4) * 4;   // K base 0,4,...,60

  const int ntiles = NUM_EDGES / 16;
  for (int t = blockIdx.x; t < ntiles; t += gridDim.x) {
    const int e0 = t * 16;

    // ---- cooperative Gaussian smearing into LDS (K padded 50->64) ----
    {
      const float d = dists[e0 + geg];
#pragma unroll
      for (int j = 0; j < 4; ++j) {
        const int k = gk0 + j;
        const float u = d - k * delta;
        sG[geg * G_LD + k] =
            (_Float16)((k < NUM_G) ? __expf(coeff * u * u) : 0.f);
      }
    }
    if (threadIdx.x < 16) {
      sC[threadIdx.x] = cutv[e0 + threadIdx.x];
      sSrc[threadIdx.x] = src[e0 + threadIdx.x];
      sDst[threadIdx.x] = dst[e0 + threadIdx.x];
    }
    __syncthreads();

    // ---- stage 1: [16x64] @ [64x16] per wave ----
    v8f acc = {};
    acc = wmma_f16(load_a_lds(sG, G_LD, 0), b1_0, acc);
    acc = wmma_f16(load_a_lds(sG, G_LD, 32), b1_1, acc);

#pragma unroll
    for (int r = 0; r < 8; ++r)
      act[(mbase + r) * LDS_LD + col] = (_Float16)ssp(acc[r] + bias1);
    __syncthreads();

    // ---- stage 2: [16x128] @ [128x16] per wave ----
    v8f acc2 = {};
#pragma unroll
    for (int ks = 0; ks < 4; ++ks)
      acc2 = wmma_f16(load_a_lds(act, LDS_LD, ks * 32), b2[ks], acc2);

    // ---- cutoff * gather * scatter-add (agg rows resolve in 192MB L2) ----
#pragma unroll
    for (int r = 0; r < 8; ++r) {
      const int m = mbase + r;
      const float wf = (acc2[r] + bias2) * sC[m];
      const float msg = wf * hW[(size_t)sSrc[m] * HID + col];
      atomicAdd(&agg[(size_t)sDst[m] * HID + col], msg);
    }
    __syncthreads();  // protect sG/act/sC before next tile overwrites
  }
}

// ---------------- kernel: h += ssp(agg @ cf_w2 + b) @ lin_w + lin_b ---------
__global__ void __launch_bounds__(256)
node_post_kernel(const float* __restrict__ agg, const float* __restrict__ cf_w2,
                 const float* __restrict__ cf_b2, const float* __restrict__ lin_w,
                 const float* __restrict__ lin_b, float* __restrict__ h) {
  __shared__ __align__(16) _Float16 act[16 * LDS_LD];
  const int lane = threadIdx.x & 31;
  const int wave = threadIdx.x >> 5;
  const int col0 = wave * 16;
  const int col = col0 + (lane & 15);

  v16h bA[4], bB[4];
#pragma unroll
  for (int ks = 0; ks < 4; ++ks) {
    bA[ks] = load_b(cf_w2, HID, ks * 32, col0);
    bB[ks] = load_b(lin_w, HID, ks * 32, col0);
  }
  const float biasA = cf_b2[col];
  const float biasB = lin_b[col];
  const int mbase = (lane >> 4) << 3;

  for (int t = blockIdx.x; t < NUM_NODES / 16; t += gridDim.x) {
    const int row0 = t * 16;
    v8f acc = {};
#pragma unroll
    for (int ks = 0; ks < 4; ++ks)
      acc = wmma_f16(load_a_g(agg, row0, ks * 32), bA[ks], acc);

#pragma unroll
    for (int r = 0; r < 8; ++r)
      act[(mbase + r) * LDS_LD + col] = (_Float16)ssp(acc[r] + biasA);
    __syncthreads();

    v8f acc2 = {};
#pragma unroll
    for (int ks = 0; ks < 4; ++ks)
      acc2 = wmma_f16(load_a_lds(act, LDS_LD, ks * 32), bB[ks], acc2);

#pragma unroll
    for (int r = 0; r < 8; ++r) {
      const size_t idx = (size_t)(row0 + mbase + r) * HID + col;
      h[idx] += acc2[r] + biasB;
    }
    __syncthreads();  // act reused next tile
  }
}

// ---------------- kernel: atom head + per-graph accumulation ----------------
__global__ void __launch_bounds__(64)
head_kernel(const float* __restrict__ h, const float* __restrict__ w1,
            const float* __restrict__ b1, const float* __restrict__ w2,
            const float* __restrict__ b2, const int* __restrict__ batch,
            float* __restrict__ gsum, float* __restrict__ gcnt) {
  __shared__ float sh[HID];
  __shared__ float sred[64];
  const int node = blockIdx.x;
  const int j = threadIdx.x;
  sh[j] = h[(size_t)node * HID + j];
  sh[j + 64] = h[(size_t)node * HID + j + 64];
  __syncthreads();
  float dot = 0.f;
#pragma unroll
  for (int k = 0; k < HID; ++k) dot += sh[k] * w1[k * 64 + j];
  sred[j] = ssp(dot + b1[j]) * w2[j];
  __syncthreads();
  if (j == 0) {
    float tot = b2[0];
#pragma unroll
    for (int k = 0; k < 64; ++k) tot += sred[k];
    const int g = batch[node];
    atomicAdd(&gsum[g], tot);
    atomicAdd(&gcnt[g], 1.0f);
  }
}

__global__ void zero_small_kernel(float* __restrict__ p, int n) {
  const int i = blockIdx.x * blockDim.x + threadIdx.x;
  if (i < n) p[i] = 0.f;
}

__global__ void finalize_kernel(const float* __restrict__ gsum,
                                const float* __restrict__ gcnt,
                                float* __restrict__ out) {
  const int g = threadIdx.x;
  if (g < NUM_GRAPHS) out[g] = gsum[g] / fmaxf(gcnt[g], 1.0f);
}

// ---------------- host launcher ----------------
extern "C" void kernel_launch(void* const* d_in, const int* in_sizes, int n_in,
                              void* d_out, int out_size, void* d_ws, size_t ws_size,
                              hipStream_t stream) {
  const float* pos      = (const float*)d_in[0];
  const float* shift    = (const float*)d_in[1];
  const float* emb      = (const float*)d_in[2];
  const float* mlp_w1   = (const float*)d_in[3];   // [3,50,128]
  const float* mlp_b1   = (const float*)d_in[4];   // [3,128]
  const float* mlp_w2   = (const float*)d_in[5];   // [3,128,128]
  const float* mlp_b2   = (const float*)d_in[6];
  const float* cf_w1    = (const float*)d_in[7];
  const float* cf_w2    = (const float*)d_in[8];
  const float* cf_b2    = (const float*)d_in[9];
  const float* lin_w    = (const float*)d_in[10];
  const float* lin_b    = (const float*)d_in[11];
  const float* head_w1  = (const float*)d_in[12];  // [128,64]
  const float* head_b1  = (const float*)d_in[13];
  const float* head_w2  = (const float*)d_in[14];  // [64,1]
  const float* head_b2  = (const float*)d_in[15];
  const int*   z        = (const int*)d_in[16];
  const int*   edge_idx = (const int*)d_in[17];    // [2,E]
  const int*   batch    = (const int*)d_in[18];
  float* out = (float*)d_out;

  const int* src = edge_idx;
  const int* dst = edge_idx + NUM_EDGES;

  // workspace layout (~90 MB of f32)
  float* ws    = (float*)d_ws;
  float* dists = ws;                                   // E
  float* cutv  = dists + NUM_EDGES;                    // E
  float* h     = cutv + NUM_EDGES;                     // N*HID
  float* hW    = h + (size_t)NUM_NODES * HID;          // N*HID
  float* agg   = hW + (size_t)NUM_NODES * HID;         // N*HID
  float* gsum  = agg + (size_t)NUM_NODES * HID;        // 128
  float* gcnt  = gsum + NUM_GRAPHS;                    // 128
  (void)ws_size; (void)n_in; (void)in_sizes; (void)out_size;

  edge_geom_kernel<<<(NUM_EDGES + 255) / 256, 256, 0, stream>>>(
      pos, shift, src, dst, dists, cutv);
  init_h_kernel<<<(NUM_NODES * HID + 255) / 256, 256, 0, stream>>>(emb, z, h);

  for (int i = 0; i < NUM_INTER; ++i) {
    const float* w1i = mlp_w1 + (size_t)i * NUM_G * HID;
    const float* b1i = mlp_b1 + (size_t)i * HID;
    const float* w2i = mlp_w2 + (size_t)i * HID * HID;
    const float* b2i = mlp_b2 + (size_t)i * HID;
    const float* cw1 = cf_w1 + (size_t)i * HID * HID;
    const float* cw2 = cf_w2 + (size_t)i * HID * HID;
    const float* cb2 = cf_b2 + (size_t)i * HID;
    const float* lw  = lin_w + (size_t)i * HID * HID;
    const float* lb  = lin_b + (size_t)i * HID;

    node_pre_kernel<<<1024, 256, 0, stream>>>(h, cw1, hW, agg);
    edge_conv_kernel<<<4096, 256, 0, stream>>>(w1i, b1i, w2i, b2i, src, dst,
                                               dists, cutv, hW, agg);
    node_post_kernel<<<1024, 256, 0, stream>>>(agg, cw2, cb2, lw, lb, h);
  }

  zero_small_kernel<<<1, 256, 0, stream>>>(gsum, 2 * NUM_GRAPHS);
  head_kernel<<<NUM_NODES, 64, 0, stream>>>(h, head_w1, head_b1, head_w2,
                                            head_b2, batch, gsum, gcnt);
  finalize_kernel<<<1, NUM_GRAPHS, 0, stream>>>(gsum, gcnt, out);
}
